// ColumnSelfAttention_2903397892820
// MI455X (gfx1250) — matrix operand order repaired and checked by reading, not compiled, not run on hardware
//
#include <hip/hip_runtime.h>
#include <hip/hip_bf16.h>

// ---------------------------------------------------------------------------
// Column self-attention for MI455X (gfx1250, wave32, WMMA).
// bf16 WMMA (v_wmma_f32_16x16x32_bf16) for all GEMMs, f32 accumulate,
// f32 softmax. V tile staged into LDS with GLOBAL_LOAD_ASYNC_TO_LDS_B128
// (ASYNCcnt), overlapped with the QK^T WMMA phase.
// ---------------------------------------------------------------------------

typedef __attribute__((ext_vector_type(16))) __bf16 v16bf;
typedef __attribute__((ext_vector_type(8)))  __bf16 v8bf;
typedef __attribute__((ext_vector_type(8)))  float  v8f;

#define RR   128          // rows (attention sequence length)
#define CC   384          // columns
#define EE   768          // embed dim
#define HH   12           // heads
#define DD   64           // head dim
#define NROW (RR * CC)    // 49152 GEMM rows (B == 1)
#define RS   (CC * EE)    // 294912: element stride between row i and i+1
#define OUT_ELEMS ((size_t)NROW * EE)              // 37,748,736 f32
#define PROB_OFF  OUT_ELEMS                        // attn_probs start in d_out

// ---------------- fragment helpers -----------------------------------------

static __device__ __forceinline__ v8f wmma_bf16(v16bf a, v16bf b, v8f c) {
  // 8-arg form: (neg_a, A, neg_b, B, c_mod, C, reuse_a, reuse_b)
  return __builtin_amdgcn_wmma_f32_16x16x32_bf16(false, a, false, b,
                                                 (short)0, c, false, false);
}

static __device__ __forceinline__ v16bf cat8(v8bf lo, v8bf hi) {
  v16bf r;
#pragma unroll
  for (int i = 0; i < 8; ++i) { r[i] = lo[i]; r[i + 8] = hi[i]; }
  return r;
}

// A-matrix 16x32 bf16 fragment. Lane layout: row M = lane&15,
// K elements half*16 .. +15 (16 contiguous) where half = lane>>4.
// `row` points at element (M, k0) of a row-major source.
static __device__ __forceinline__ v16bf loadA_bf16(const __bf16* row, int half) {
  const v8bf* p0 = (const v8bf*)(row + half * 16);
  const v8bf* p1 = (const v8bf*)(row + half * 16 + 8);
  return cat8(p0[0], p1[0]);
}

// Same fragment but sourced from f32 memory (converted in-register).
static __device__ __forceinline__ v16bf loadA_f32(const float* row, int half) {
  float f[16];
  const float4* p0 = (const float4*)(row + half * 16);
  const float4* p1 = (const float4*)(row + half * 16 + 8);
  float4 a = p0[0], b = p0[1], c = p1[0], d = p1[1];
  f[0]=a.x; f[1]=a.y; f[2]=a.z;  f[3]=a.w;  f[4]=b.x;  f[5]=b.y;  f[6]=b.z;  f[7]=b.w;
  f[8]=c.x; f[9]=c.y; f[10]=c.z; f[11]=c.w; f[12]=d.x; f[13]=d.y; f[14]=d.z; f[15]=d.w;
  v16bf r;
#pragma unroll
  for (int i = 0; i < 16; ++i) r[i] = (__bf16)f[i];
  return r;
}

// B-matrix 32x16 bf16 fragment. Lane holds column N = lane&15, K elements
// k0 + half*16 .. +15 (16 contiguous). `p` points at that first element.
static __device__ __forceinline__ v16bf loadB16(const __bf16* p) {
  const v8bf* q = (const v8bf*)p;
  return cat8(q[0], q[1]);
}

// ---------------- weight conversion -----------------------------------------

__global__ void cvt_f32_to_bf16(const float* __restrict__ in,
                                __bf16* __restrict__ out, int n) {
  int i = blockIdx.x * blockDim.x + threadIdx.x;
  if (i < n) out[i] = (__bf16)in[i];
}

// ---------------- fused QKV projection ---------------------------------------
// grid (192, 6, 3), block 256 (8 waves, 4x2 of 64x64 tiles). Out = x @ W^T + b
// (scaled for Q). A read from f32 x, converted in-register; B from bf16 W.

__global__ __launch_bounds__(256) void qkv_gemm(
    const float* __restrict__ x,
    const __bf16* __restrict__ wq, const __bf16* __restrict__ wk,
    const __bf16* __restrict__ wv,
    const float* __restrict__ bq, const float* __restrict__ bk,
    const float* __restrict__ bv,
    __bf16* __restrict__ Q, __bf16* __restrict__ K, __bf16* __restrict__ V) {
  const int which = blockIdx.z;
  const __bf16* W   = (which == 0) ? wq : (which == 1) ? wk : wv;
  const float*  bia = (which == 0) ? bq : (which == 1) ? bk : bv;
  __bf16*       out = (which == 0) ? Q  : (which == 1) ? K  : V;
  const float scale = (which == 0) ? 0.125f : 1.0f;  // D^-0.5 = 1/8

  const int tid = threadIdx.x, wave = tid >> 5, lane = tid & 31;
  const int lrow = lane & 15, half = lane >> 4;
  const int waveM = blockIdx.x * 256 + (wave >> 1) * 64;
  const int waveN = blockIdx.y * 128 + (wave & 1) * 64;

  v8f acc[4][4];
#pragma unroll
  for (int i = 0; i < 4; ++i)
#pragma unroll
    for (int j = 0; j < 4; ++j) acc[i][j] = (v8f)0.0f;

  for (int k0 = 0; k0 < EE; k0 += 32) {
    v16bf bfr[4];
#pragma unroll
    for (int bn = 0; bn < 4; ++bn)
      bfr[bn] = loadB16(W + (size_t)(waveN + 16 * bn + lrow) * EE + k0 + half * 16);
#pragma unroll
    for (int am = 0; am < 4; ++am) {
      v16bf afr = loadA_f32(x + (size_t)(waveM + 16 * am + lrow) * EE + k0, half);
#pragma unroll
      for (int bn = 0; bn < 4; ++bn)
        acc[am][bn] = wmma_bf16(afr, bfr[bn], acc[am][bn]);
    }
  }

#pragma unroll
  for (int am = 0; am < 4; ++am)
#pragma unroll
    for (int bn = 0; bn < 4; ++bn) {
      const int col = waveN + 16 * bn + lrow;
      const float bv_ = bia[col];
#pragma unroll
      for (int r = 0; r < 8; ++r) {
        const int row = waveM + 16 * am + half * 8 + r;
        out[(size_t)row * EE + col] = (__bf16)((acc[am][bn][r] + bv_) * scale);
      }
    }
}

// ---------------- attention per (column c, head h) ---------------------------
// grid (384, 12), block 256 (8 waves), dynamic LDS 136,832 B:
//   [0,      67584)  S  : f32 logits, row stride 132 f32 (528 B)
//   [67584, 102400)  P  : bf16 probs, row stride 136 bf16 (272 B)
//   [102400,119808)  Vt : V transposed bf16, Vt[d][j], row stride 136 bf16
//   [119808,136192)  Vrm: V row-major bf16 (async-copy landing zone)
//   [136192,136320)  Mk : key padding mask bytes
//   [136320,136832)  inv: per-row 1/sum

#define S_OFF    0
#define P_OFF    67584
#define VT_OFF   102400
#define VRM_OFF  119808
#define MK_OFF   136192
#define INV_OFF  136320
#define ATTN_LDS 136832

__global__ __launch_bounds__(256) void attn_kernel(
    const __bf16* __restrict__ Q, const __bf16* __restrict__ Kb,
    const __bf16* __restrict__ V, const unsigned char* __restrict__ mask,
    float* __restrict__ probs_out, __bf16* __restrict__ ctx) {
  extern __shared__ char smem[];
  const int c = blockIdx.x, h = blockIdx.y;
  const int tid = threadIdx.x, wave = tid >> 5, lane = tid & 31;
  const int lrow = lane & 15, half = lane >> 4;
  const size_t base = (size_t)c * EE + h * DD;

  float*  Sf   = (float*)(smem + S_OFF);
  __bf16* Pb   = (__bf16*)(smem + P_OFF);
  __bf16* Vt   = (__bf16*)(smem + VT_OFF);
  __bf16* Vrm  = (__bf16*)(smem + VRM_OFF);
  unsigned char* Mk = (unsigned char*)(smem + MK_OFF);
  float*  invp = (float*)(smem + INV_OFF);

  // mask column (normal load) + async-stream V tile row-major into LDS.
  // 1024 x 16B chunks; chunk q = global row q/8, bytes (q%8)*16 within row,
  // landing contiguously at VRM_OFF + q*16. Overlaps with the QK^T phase.
  if (tid < RR) Mk[tid] = mask[(size_t)tid * CC + c];
#pragma unroll
  for (int it = 0; it < 4; ++it) {
    const int q = tid + it * 256;
    const int j = q >> 3;
    const __bf16* gp = V + (size_t)j * RS + base + (q & 7) * 8;
    const unsigned ldsoff = VRM_OFF + q * 16;
    asm volatile("global_load_async_to_lds_b128 %0, %1, off"
                 :: "v"(ldsoff), "v"(gp) : "memory");
  }

  // ---- S = Q K^T : wave w owns columns [16w, 16w+16), all 128 rows ----
  v8f sacc[8];
#pragma unroll
  for (int mt = 0; mt < 8; ++mt) sacc[mt] = (v8f)0.0f;
#pragma unroll
  for (int d0 = 0; d0 < DD; d0 += 32) {
    v16bf bfr = loadB16(Kb + (size_t)(16 * wave + lrow) * RS + base + d0 + half * 16);
#pragma unroll
    for (int mt = 0; mt < 8; ++mt) {
      v16bf afr = loadA_bf16(Q + (size_t)(16 * mt + lrow) * RS + base + d0, half);
      sacc[mt] = wmma_bf16(afr, bfr, sacc[mt]);
    }
  }
#pragma unroll
  for (int mt = 0; mt < 8; ++mt)
#pragma unroll
    for (int r = 0; r < 8; ++r)
      Sf[(16 * mt + half * 8 + r) * 132 + 16 * wave + lrow] = sacc[mt][r];

  asm volatile("s_wait_asynccnt 0x0" ::: "memory");  // V tile landed
  __syncthreads();

  // ---- concurrently: transpose V into B-fragment layout + row reductions ----
  for (int idx = tid; idx < RR * DD; idx += 256) {
    const int j = idx >> 6, d = idx & 63;
    Vt[d * 136 + j] = Vrm[idx];
  }
  if (tid < RR) {
    const int i = tid;
    float* Srow = Sf + (size_t)i * 132;
    float mx = -3.4e38f;
    for (int j = 0; j < RR; ++j) {
      const float v = Mk[j] ? -10000.0f : Srow[j];
      mx = fmaxf(mx, v);
    }
    float sum = 0.0f;
    for (int j = 0; j < RR; ++j) {
      const float v = Mk[j] ? -10000.0f : Srow[j];
      const float e = expf(v - mx);
      Srow[j] = e;                 // keep exp() for the cooperative pass
      sum += e;
    }
    invp[i] = 1.0f / sum;
  }
  __syncthreads();

  // ---- cooperative, fully-coalesced probs store + bf16 P for the PV GEMM ----
  float* pg = probs_out + (size_t)(h * CC + c) * (RR * RR);
  for (int idx = tid; idx < RR * RR; idx += 256) {
    const int i = idx >> 7, j = idx & 127;
    const float p = Sf[i * 132 + j] * invp[i];
    pg[idx] = p;
    Pb[i * 136 + j] = (__bf16)p;
  }
  __syncthreads();

  // ---- context = P @ V : wave w owns rows [16w, 16w+16), all 64 cols ----
  v8f cacc[4];
#pragma unroll
  for (int nt = 0; nt < 4; ++nt) cacc[nt] = (v8f)0.0f;
#pragma unroll
  for (int j0 = 0; j0 < RR; j0 += 32) {
    v16bf afr = loadA_bf16(Pb + (size_t)(16 * wave + lrow) * 136 + j0, half);
#pragma unroll
    for (int nt = 0; nt < 4; ++nt) {
      v16bf bfr = loadB16(Vt + (size_t)(16 * nt + lrow) * 136 + j0 + half * 16);
      cacc[nt] = wmma_bf16(afr, bfr, cacc[nt]);
    }
  }
#pragma unroll
  for (int nt = 0; nt < 4; ++nt)
#pragma unroll
    for (int r = 0; r < 8; ++r) {
      const int i = 16 * wave + half * 8 + r;
      const int d = 16 * nt + lrow;
      ctx[(size_t)((size_t)i * CC + c) * EE + h * DD + d] = (__bf16)cacc[nt][r];
    }
}

// ---------------- output projection ------------------------------------------
// grid (192, 6), block 256. out = ctx @ wo^T + bo, f32 result.

__global__ __launch_bounds__(256) void out_gemm(
    const __bf16* __restrict__ ctx, const __bf16* __restrict__ wo,
    const float* __restrict__ bo, float* __restrict__ out) {
  const int tid = threadIdx.x, wave = tid >> 5, lane = tid & 31;
  const int lrow = lane & 15, half = lane >> 4;
  const int waveM = blockIdx.x * 256 + (wave >> 1) * 64;
  const int waveN = blockIdx.y * 128 + (wave & 1) * 64;

  v8f acc[4][4];
#pragma unroll
  for (int i = 0; i < 4; ++i)
#pragma unroll
    for (int j = 0; j < 4; ++j) acc[i][j] = (v8f)0.0f;

  for (int k0 = 0; k0 < EE; k0 += 32) {
    v16bf bfr[4];
#pragma unroll
    for (int bn = 0; bn < 4; ++bn)
      bfr[bn] = loadB16(wo + (size_t)(waveN + 16 * bn + lrow) * EE + k0 + half * 16);
#pragma unroll
    for (int am = 0; am < 4; ++am) {
      v16bf afr = loadA_bf16(ctx + (size_t)(waveM + 16 * am + lrow) * EE + k0, half);
#pragma unroll
      for (int bn = 0; bn < 4; ++bn)
        acc[am][bn] = wmma_bf16(afr, bfr[bn], acc[am][bn]);
    }
  }

#pragma unroll
  for (int am = 0; am < 4; ++am)
#pragma unroll
    for (int bn = 0; bn < 4; ++bn) {
      const int col = waveN + 16 * bn + lrow;
      const float bv_ = bo[col];
#pragma unroll
      for (int r = 0; r < 8; ++r) {
        const int row = waveM + 16 * am + half * 8 + r;
        out[(size_t)row * EE + col] = acc[am][bn][r] + bv_;
      }
    }
}

// ---------------- launcher ---------------------------------------------------

extern "C" void kernel_launch(void* const* d_in, const int* in_sizes, int n_in,
                              void* d_out, int out_size, void* d_ws, size_t ws_size,
                              hipStream_t stream) {
  const float* x  = (const float*)d_in[0];
  // d_in[1] = distances (unused)
  const unsigned char* mask = (const unsigned char*)d_in[2];  // bool (B,R,C)
  const float* wq = (const float*)d_in[3];
  const float* bq = (const float*)d_in[4];
  const float* wk = (const float*)d_in[5];
  const float* bk = (const float*)d_in[6];
  const float* wv = (const float*)d_in[7];
  const float* bv = (const float*)d_in[8];
  const float* wo = (const float*)d_in[9];
  const float* bo = (const float*)d_in[10];
  float* out = (float*)d_out;

  // workspace layout (bytes), all 256-aligned; total ~307 MB
  char* ws = (char*)d_ws;
  const size_t WB = (size_t)EE * EE * 2;            // 1,179,648
  const size_t QB = (size_t)NROW * EE * 2;          // 75,497,472
  __bf16* wq_bf = (__bf16*)(ws + 0 * WB);
  __bf16* wk_bf = (__bf16*)(ws + 1 * WB);
  __bf16* wv_bf = (__bf16*)(ws + 2 * WB);
  __bf16* wo_bf = (__bf16*)(ws + 3 * WB);
  __bf16* Qbf   = (__bf16*)(ws + 4 * WB);
  __bf16* Kbf   = (__bf16*)(ws + 4 * WB + QB);
  __bf16* Vbf   = (__bf16*)(ws + 4 * WB + 2 * QB);
  __bf16* Cbf   = (__bf16*)(ws + 4 * WB + 3 * QB);

  const int WN = EE * EE;                           // 589824
  const int cvtBlocks = (WN + 255) / 256;
  cvt_f32_to_bf16<<<cvtBlocks, 256, 0, stream>>>(wq, wq_bf, WN);
  cvt_f32_to_bf16<<<cvtBlocks, 256, 0, stream>>>(wk, wk_bf, WN);
  cvt_f32_to_bf16<<<cvtBlocks, 256, 0, stream>>>(wv, wv_bf, WN);
  cvt_f32_to_bf16<<<cvtBlocks, 256, 0, stream>>>(wo, wo_bf, WN);

  qkv_gemm<<<dim3(NROW / 256, EE / 128, 3), 256, 0, stream>>>(
      x, wq_bf, wk_bf, wv_bf, bq, bk, bv, Qbf, Kbf, Vbf);

  attn_kernel<<<dim3(CC, HH), 256, ATTN_LDS, stream>>>(
      Qbf, Kbf, Vbf, mask, out + PROB_OFF, Cbf);

  out_gemm<<<dim3(NROW / 256, EE / 128), 256, 0, stream>>>(Cbf, wo_bf, bo, out);
}
